// AttentionHead_51531017617866
// MI455X (gfx1250) — matrix operand order repaired and checked by reading, not compiled, not run on hardware
//
#include <hip/hip_runtime.h>

// ---------------------------------------------------------------------------
// Single attention head (d=128) for MI455X / gfx1250, wave32 + WMMA bf16.
//   Kernel 1: Q/K/V = x @ W^T  (bf16 WMMA, f32 accum), LDS-staged bf16 tiles,
//             fused RoPE on Q,K; V stored transposed [B,128,S].
//   Kernel 2: causal flash attention; K and V^T tiles staged into LDS with
//             global_load_async_to_lds_b128 (ASYNCcnt), WMMA scores + PV.
// ---------------------------------------------------------------------------

typedef __attribute__((ext_vector_type(16))) __bf16 v16bf;
typedef __attribute__((ext_vector_type(8)))  __bf16 bf16x8;
typedef __attribute__((ext_vector_type(8)))  float  v8f;
typedef __attribute__((ext_vector_type(4)))  float  f32x4;

#define S_LEN   4096
#define D_MODEL 4096
#define HEAD    128
#define NEG_BIG (-3.0e38f)

static __device__ __forceinline__ v16bf make_frag(bf16x8 lo, bf16x8 hi) {
  v16bf r;
#pragma unroll
  for (int i = 0; i < 8; ++i) { r[i] = lo[i]; r[i + 8] = hi[i]; }
  return r;
}

// LDS byte offset of a pointer into a __shared__ array (addrspace(3) cast).
typedef __attribute__((address_space(3))) const char lds_char_t;
static __device__ __forceinline__ unsigned lds_addr_of(const void* p) {
  return (unsigned)(size_t)(lds_char_t*)p;
}

// CDNA5 async copy: LDS[lds_byte] = MEM[gptr .. gptr+15], tracked by ASYNCcnt.
static __device__ __forceinline__ void async_copy_b128(unsigned lds_byte,
                                                       const void* gptr) {
  asm volatile("global_load_async_to_lds_b128 %0, %1, off"
               :: "v"(lds_byte), "v"(gptr)
               : "memory");
}
static __device__ __forceinline__ void wait_async0() {
  asm volatile("s_wait_asynccnt 0" ::: "memory");
}

// ---------------------------------------------------------------------------
// Kernel 1: QKV projection + RoPE.
// grid = (M/128, 3), block = 256 (8 waves). blockIdx.y selects Q/K/V.
// Per 32-wide K-step: 256 threads cooperatively convert the 128x32 x-tile and
// the 128x32 W-tile to bf16 in LDS (one conversion instead of per-wave
// redundant ones); fragments are then pure ds_load_b128 feeding WMMA.
// ---------------------------------------------------------------------------
__global__ void __launch_bounds__(256)
qkv_rope_kernel(const float* __restrict__ x,
                const float* __restrict__ Wq,
                const float* __restrict__ Wk,
                const float* __restrict__ Wv,
                __bf16* __restrict__ Qb,
                __bf16* __restrict__ Kb,
                __bf16* __restrict__ Vt) {
  __shared__ __align__(16) __bf16 xs[128 * 32];   // x tile  [row][k]  8 KB
  __shared__ __align__(16) __bf16 wt[128 * 32];   // W tile  [col][k]  8 KB

  const int which = blockIdx.y;
  const float* W  = (which == 0) ? Wq : (which == 1) ? Wk : Wv;

  const int tid  = threadIdx.x;
  const int wave = tid >> 5;
  const int lane = tid & 31;
  const int half = lane >> 4;
  const int l16  = lane & 15;
  const int hsel = half * 8;

  // staging assignment: each thread owns one 16-wide half-row of both tiles
  const int srow = tid >> 1;            // 0..127
  const int sko  = (tid & 1) * 16;      // 0 or 16

  v8f acc[8];
#pragma unroll
  for (int g = 0; g < 8; ++g) acc[g] = (v8f)0.f;

  for (int kb = 0; kb < D_MODEL; kb += 32) {
    __syncthreads();
    {
      // ---- stage x tile (fp32 -> bf16) ----
      const float* xp = x + (size_t)(blockIdx.x * 128 + srow) * D_MODEL + kb + sko;
      f32x4 x0 = *(const f32x4*)(xp);
      f32x4 x1 = *(const f32x4*)(xp + 4);
      f32x4 x2 = *(const f32x4*)(xp + 8);
      f32x4 x3 = *(const f32x4*)(xp + 12);
      bf16x8 c0, c1;
#pragma unroll
      for (int i = 0; i < 4; ++i) {
        c0[i] = (__bf16)x0[i]; c0[i + 4] = (__bf16)x1[i];
        c1[i] = (__bf16)x2[i]; c1[i + 4] = (__bf16)x3[i];
      }
      *(bf16x8*)&xs[srow * 32 + sko]     = c0;
      *(bf16x8*)&xs[srow * 32 + sko + 8] = c1;
      if (kb + 32 < D_MODEL) __builtin_prefetch(xp + 32, 0, 1);

      // ---- stage W tile (fp32 -> bf16) ----
      const float* wp = W + (size_t)srow * D_MODEL + kb + sko;
      f32x4 w0 = *(const f32x4*)(wp);
      f32x4 w1 = *(const f32x4*)(wp + 4);
      f32x4 w2 = *(const f32x4*)(wp + 8);
      f32x4 w3 = *(const f32x4*)(wp + 12);
#pragma unroll
      for (int i = 0; i < 4; ++i) {
        c0[i] = (__bf16)w0[i]; c0[i + 4] = (__bf16)w1[i];
        c1[i] = (__bf16)w2[i]; c1[i + 4] = (__bf16)w3[i];
      }
      *(bf16x8*)&wt[srow * 32 + sko]     = c0;
      *(bf16x8*)&wt[srow * 32 + sko + 8] = c1;
      if (kb + 32 < D_MODEL) __builtin_prefetch(wp + 32, 0, 1);
    }
    __syncthreads();

    // ---- A fragment from LDS (16x32 bf16; ISA A-layout) ----
    const __bf16* ap = &xs[(wave * 16 + l16) * 32 + hsel];
    v16bf afrag = make_frag(*(const bf16x8*)(ap), *(const bf16x8*)(ap + 16));

    // ---- 8 column groups: B fragment from LDS + WMMA ----
#pragma unroll
    for (int g = 0; g < 8; ++g) {
      const __bf16* bp = &wt[(g * 16 + l16) * 32 + half * 8];
      v16bf bfrag = make_frag(*(const bf16x8*)(bp), *(const bf16x8*)(bp + 16));
      acc[g] = __builtin_amdgcn_wmma_f32_16x16x32_bf16(
          false, afrag, false, bfrag, (short)0, acc[g], false, false);
    }
  }

  // ---- RoPE (pairs = adjacent lanes in C layout) + store ----
#pragma unroll
  for (int g = 0; g < 8; ++g) {
    const int n = g * 16 + l16;
#pragma unroll
    for (int j = 0; j < 8; ++j) {
      const int r = blockIdx.x * 128 + wave * 16 + j + half * 8;  // b*S + s
      float v = acc[g][j];
      if (which < 2) {
        const int   s     = r & (S_LEN - 1);
        const int   p     = n >> 1;
        const float theta = __expf((float)p * -0.28782313662f);  // 10000^(-p/32)
        float c, sn;
        __sincosf((float)s * theta, &sn, &c);
        const float other = __shfl_xor(v, 1, 32);
        v = (n & 1) ? (v * c + other * sn) : (v * c - other * sn);
      }
      if (which == 0)      Qb[(size_t)r * HEAD + n] = (__bf16)v;
      else if (which == 1) Kb[(size_t)r * HEAD + n] = (__bf16)v;
      else {               // V stored transposed: [B, 128, S]
        const int b = r >> 12, s = r & (S_LEN - 1);
        Vt[((size_t)b * HEAD + n) * S_LEN + s] = (__bf16)v;
      }
    }
  }
}

// ---------------------------------------------------------------------------
// Kernel 2: causal flash attention.
// grid = (S/128, B), block = 256 (8 waves); each wave owns 16 query rows.
// Per 32-key tile: async-stage K[32][128] and V^T rows [128][32] into LDS,
// 8 score WMMAs, online softmax, P via per-wave LDS (C->A layout), 8 PV WMMAs.
// ---------------------------------------------------------------------------
__global__ void __launch_bounds__(256)
flash_attn_kernel(const __bf16* __restrict__ Qb,
                  const __bf16* __restrict__ Kb,
                  const __bf16* __restrict__ Vt,
                  float* __restrict__ outp) {
  __shared__ __align__(16) __bf16 Ks [32 * 128];   // K tile, memory layout (8 KB)
  __shared__ __align__(16) __bf16 Vts[128 * 40];   // V^T tile [h][key], padded
  __shared__ __align__(16) __bf16 Pbuf[8 * 16 * 32];

  const int qtile = blockIdx.x;
  const int batch = blockIdx.y;
  const int tid   = threadIdx.x;
  const int wave  = tid >> 5;
  const int lane  = tid & 31;
  const int half  = lane >> 4;
  const int l16   = lane & 15;
  const size_t base = (size_t)batch * S_LEN;

  const unsigned ks_base  = lds_addr_of(&Ks[0]);
  const unsigned vts_base = lds_addr_of(&Vts[0]);

  // ---- Q fragments held in registers ----
  v16bf aQ[4];
  {
    const int qrow = qtile * 128 + wave * 16 + l16;
    const __bf16* qp = Qb + (base + qrow) * HEAD + half * 8;
#pragma unroll
    for (int t = 0; t < 4; ++t)
      aQ[t] = make_frag(*(const bf16x8*)(qp + t * 32),
                        *(const bf16x8*)(qp + t * 32 + 16));
  }

  v8f O[8];
#pragma unroll
  for (int g = 0; g < 8; ++g) O[g] = (v8f)0.f;
  float m[8], l[8];
#pragma unroll
  for (int j = 0; j < 8; ++j) { m[j] = NEG_BIG; l[j] = 0.f; }

  const float scale  = 0.08838834764832f;  // 1/sqrt(128)
  const int   ntiles = qtile * 4 + 4;      // causal bound, uniform per block
  __bf16* Pw = &Pbuf[wave * 16 * 32];

  for (int t = 0; t < ntiles; ++t) {
    const int kb = t * 32;

    // ---- async stage: K tile (8 KB, straight copy) -----------------------
    __syncthreads();
    {
      const __bf16* ksrc = Kb + (base + kb) * HEAD;     // 32 rows x 256 B
#pragma unroll
      for (int rnd = 0; rnd < 2; ++rnd) {
        const int c = rnd * 256 + tid;                  // 16-byte chunk id
        async_copy_b128(ks_base + c * 16, ksrc + c * 8);
      }
      // ---- async stage: V^T rows (64 B per h-row, padded to 80 B) -------
#pragma unroll
      for (int rnd = 0; rnd < 2; ++rnd) {
        const int c  = rnd * 256 + tid;                 // 0..511
        const int h  = c >> 2;                          // 0..127
        const int cc = c & 3;                           // 16-B chunk in row
        const __bf16* vsrc =
            Vt + ((size_t)(batch * HEAD + h)) * S_LEN + kb + cc * 8;
        async_copy_b128(vts_base + h * 80 + cc * 16, vsrc);
      }
      wait_async0();
    }
    __syncthreads();

    // ---- scores: S = Q @ K^T (B-frags from LDS K tile) -------------------
    v8f sc0 = (v8f)0.f, sc1 = (v8f)0.f;
#pragma unroll
    for (int tt = 0; tt < 4; ++tt) {
      const __bf16* kp0 = &Ks[l16 * 128 + tt * 32 + half * 8];
      const __bf16* kp1 = kp0 + 16 * 128;
      sc0 = __builtin_amdgcn_wmma_f32_16x16x32_bf16(
          false, aQ[tt], false,
          make_frag(*(const bf16x8*)(kp0), *(const bf16x8*)(kp0 + 16)),
          (short)0, sc0, false, false);
      sc1 = __builtin_amdgcn_wmma_f32_16x16x32_bf16(
          false, aQ[tt], false,
          make_frag(*(const bf16x8*)(kp1), *(const bf16x8*)(kp1 + 16)),
          (short)0, sc1, false, false);
    }

    // ---- online softmax --------------------------------------------------
    const int key0 = kb + l16;
#pragma unroll
    for (int j = 0; j < 8; ++j) {
      const int gq = qtile * 128 + wave * 16 + j + half * 8;
      float s0 = sc0[j] * scale;
      float s1 = sc1[j] * scale;
      if (key0 > gq)      s0 = NEG_BIG;
      if (key0 + 16 > gq) s1 = NEG_BIG;

      float loc = fmaxf(s0, s1);
#pragma unroll
      for (int o = 1; o < 16; o <<= 1) loc = fmaxf(loc, __shfl_xor(loc, o, 32));
      const float mnew  = fmaxf(m[j], loc);
      const float alpha = __expf(m[j] - mnew);
      m[j] = mnew;

      const float p0 = __expf(s0 - mnew);
      const float p1 = __expf(s1 - mnew);
      float rs = p0 + p1;
#pragma unroll
      for (int o = 1; o < 16; o <<= 1) rs += __shfl_xor(rs, o, 32);
      l[j] = l[j] * alpha + rs;
#pragma unroll
      for (int g = 0; g < 8; ++g) O[g][j] *= alpha;

      const int r = j + half * 8;
      Pw[r * 32 + l16]      = (__bf16)p0;
      Pw[r * 32 + 16 + l16] = (__bf16)p1;
    }

    // ---- aP (A-layout) from per-wave LDS; same-wave DS ops are in-order --
    const __bf16* pp = Pw + l16 * 32 + half * 8;
    v16bf aP = make_frag(*(const bf16x8*)(pp), *(const bf16x8*)(pp + 16));

    // ---- O += P @ V  (B-frag = contiguous run in V^T LDS row) ------------
#pragma unroll
    for (int g = 0; g < 8; ++g) {
      const __bf16* vtp = &Vts[(g * 16 + l16) * 40 + half * 8];
      O[g] = __builtin_amdgcn_wmma_f32_16x16x32_bf16(
          false, aP, false,
          make_frag(*(const bf16x8*)(vtp), *(const bf16x8*)(vtp + 16)),
          (short)0, O[g], false, false);
    }
  }

  // ---- epilogue: normalize, store fp32 ----
#pragma unroll
  for (int g = 0; g < 8; ++g) {
#pragma unroll
    for (int j = 0; j < 8; ++j) {
      const int r = qtile * 128 + wave * 16 + j + half * 8;
      outp[(base + r) * HEAD + g * 16 + l16] = O[g][j] / l[j];
    }
  }
}

// ---------------------------------------------------------------------------
extern "C" void kernel_launch(void* const* d_in, const int* in_sizes, int n_in,
                              void* d_out, int out_size, void* d_ws, size_t ws_size,
                              hipStream_t stream) {
  const float* x  = (const float*)d_in[0];   // [2, 4096, 4096]
  const float* Wq = (const float*)d_in[1];   // [128, 4096]
  const float* Wk = (const float*)d_in[2];
  const float* Wv = (const float*)d_in[3];

  // workspace: Q,K [B*S,128] bf16; V^T [B,128,S] bf16 (2 MB each, 6 MB total)
  __bf16* Qb = (__bf16*)d_ws;
  __bf16* Kb = Qb + (size_t)8192 * HEAD;
  __bf16* Vt = Kb + (size_t)8192 * HEAD;

  qkv_rope_kernel<<<dim3(64, 3), 256, 0, stream>>>(x, Wq, Wk, Wv, Qb, Kb, Vt);
  flash_attn_kernel<<<dim3(32, 2), 256, 0, stream>>>(Qb, Kb, Vt, (float*)d_out);
}
